// ProjectedGaussianRasterizer_7421703487871
// MI455X (gfx1250) — matrix operand order repaired and checked by reading, not compiled, not run on hardware
//
#include <hip/hip_runtime.h>

typedef __attribute__((ext_vector_type(16))) _Float16 v16h;
typedef __attribute__((ext_vector_type(8)))  float    v8f;

// b128 async-copy payload type: int4 pointers in explicit address spaces.
typedef int v4i __attribute__((vector_size(16)));
typedef __attribute__((address_space(1))) v4i gv4i;   // global
typedef __attribute__((address_space(3))) v4i lv4i;   // LDS

#define G_TOT     512
#define G_STRIDE  10
#define ALPHA_THR (1.0f/255.0f)
#define TRANS_THR 1e-4f
#define ALPHA_CAP 0.99f

#define HAS_ASYNC_LDS __has_builtin(__builtin_amdgcn_global_load_async_to_lds_b128)

// ---------------------------------------------------------------------------
// Kernel 1: stable depth rank + pack sorted gaussians (AoS, 10 f32 each)
// packed[g] = { mx, my, A, B, C, opacity, r, g, b, pad }
// ---------------------------------------------------------------------------
__global__ __launch_bounds__(512)
void sort_pack_kernel(const float* __restrict__ means,
                      const float* __restrict__ conics,
                      const float* __restrict__ colors,
                      const float* __restrict__ opac,
                      const float* __restrict__ depths,
                      float* __restrict__ packed) {
  __shared__ float d[G_TOT];
  int i = threadIdx.x;
  d[i] = depths[i];
  __syncthreads();
  float di = d[i];
  int rank = 0;
  for (int j = 0; j < G_TOT; ++j) {
    float dj = d[j];
    // stable: ties broken by original index (matches stable argsort)
    rank += (dj < di) || ((dj == di) && (j < i));
  }
  float* o = packed + rank * G_STRIDE;
  o[0] = means[2*i+0];
  o[1] = means[2*i+1];
  o[2] = conics[3*i+0];
  o[3] = conics[3*i+1];
  o[4] = conics[3*i+2];
  o[5] = opac[i];
  o[6] = colors[3*i+0];
  o[7] = colors[3*i+1];
  o[8] = colors[3*i+2];
  o[9] = 0.0f;
}

__device__ __forceinline__ float eval_alpha(const float* gs, int g,
                                            float px, float py) {
  const float* q = gs + g * G_STRIDE;      // broadcast LDS reads
  float dx  = px - q[0];
  float dy  = py - q[1];
  float sig = 0.5f * (q[2]*dx*dx + q[4]*dy*dy) + q[3]*dx*dy;
  float a   = fminf(ALPHA_CAP, q[5] * __expf(-sig));
  return ((sig >= 0.0f) && (a >= ALPHA_THR)) ? a : 0.0f;
}

// ---------------------------------------------------------------------------
// Kernel 2: rasterize. 8 waves/block; each wave owns 16 consecutive pixels.
// Gaussian table staged into LDS via async global->LDS (ASYNCcnt-tracked).
// Per 32-gaussian chunk: contribution matrix A (16x32 f16) via cross-lane
// transmittance scan, colors as B (32x16 f16, cols 0..2 = RGB), accumulated
// with v_wmma_f32_16x16x32_f16.
// ---------------------------------------------------------------------------
__global__ __launch_bounds__(256)
void raster_kernel(const float* __restrict__ packed, float* __restrict__ out) {
  __shared__ __align__(16) float gs[G_TOT * G_STRIDE]; // 20480 B: all gaussians
  __shared__ float dscr[8][256];                       // per-wave 16x16 D tile

  int tid = threadIdx.x;
#if HAS_ASYNC_LDS
  // 20480 B = 1280 x b128 chunks; 256 lanes -> 5 async issues per thread
  for (int i = tid * 4; i < G_TOT * G_STRIDE; i += 256 * 4) {
    __builtin_amdgcn_global_load_async_to_lds_b128(
        (gv4i*)(packed + i), (lv4i*)(&gs[i]), /*offset=*/0, /*cpol=*/0);
  }
#if __has_builtin(__builtin_amdgcn_s_wait_asynccnt)
  __builtin_amdgcn_s_wait_asynccnt(0);
#else
  asm volatile("s_wait_asynccnt 0" ::: "memory");
#endif
  __syncthreads();
#else
  for (int i = tid; i < G_TOT * G_STRIDE; i += 256) gs[i] = packed[i];
  __syncthreads();
#endif

  int  wave = tid >> 5;
  int  lane = tid & 31;
  int  m    = lane & 15;           // pixel-in-strip == A-matrix row == B col N
  bool hi   = lane >= 16;

  int   p  = blockIdx.x * 128 + wave * 16 + m;  // linear pixel id (256x256)
  float px = (float)(p & 255) + 0.5f;
  float py = (float)(p >> 8)  + 0.5f;

  float T   = 1.0f;                // transmittance before next gaussian
  v8f   acc = {};                  // 16x16 f32 accumulator (cols 0..2 = RGB)

  // ISA 16-bit A layout: lanes 0-15 hold K 0-7 & 16-23; lanes 16-31 hold
  // K 8-15 & 24-31 (per pixel row). B layout: same K order as A,
  // N = lane&15.
  const int koffA = hi ? 8  : 0;
  const int koffB = hi ? 24 : 16;
  const int bkoff = hi ? 16 : 0;

  for (int base = 0; base < G_TOT; base += 32) {
    float al[16];
    float qA = 1.0f, qB = 1.0f;    // segment transparency products
#pragma unroll
    for (int e = 0; e < 8; ++e) {
      al[e]     = eval_alpha(gs, base + koffA + e, px, py);
      qA       *= (1.0f - al[e]);
      al[8 + e] = eval_alpha(gs, base + koffB + e, px, py);
      qB       *= (1.0f - al[8 + e]);
    }
    // stitch cumulative product across the lane pair owning this pixel
    float qAo = __shfl_xor(qA, 16, 32);
    float qBo = __shfl_xor(qB, 16, 32);
    float tA  = hi ? T * qAo : T;                       // T before my seg A
    float tB  = T * qA * qAo * (hi ? qBo : 1.0f);       // T before my seg B

    v16h Am;
    float t = tA;
#pragma unroll
    for (int e = 0; e < 8; ++e) {
      float c = (t > TRANS_THR) ? al[e] * t : 0.0f;     // live-gated contrib
      Am[e]   = (_Float16)c;
      t      *= (1.0f - al[e]);
    }
    t = tB;
#pragma unroll
    for (int e = 0; e < 8; ++e) {
      float c   = (t > TRANS_THR) ? al[8 + e] * t : 0.0f;
      Am[8 + e] = (_Float16)c;
      t        *= (1.0f - al[8 + e]);
    }
    T *= qA * qB * qAo * qBo;

    v16h Bm;
#pragma unroll
    for (int e = 0; e < 16; ++e) {
      float cv = (m < 3) ? gs[(base + (e < 8 ? koffA + e : koffB + e - 8)) * G_STRIDE + 6 + m] : 0.0f;
      Bm[e] = (_Float16)cv;
    }

    acc = __builtin_amdgcn_wmma_f32_16x16x32_f16(
        /*neg_a=*/false, Am, /*neg_b=*/false, Bm,
        /*c_mod=*/(short)0, acc, /*reuse_a=*/false, /*reuse_b=*/false);

    // wave-uniform early termination (T monotone => future contribs all 0)
    float tmax = T;
#pragma unroll
    for (int s = 1; s < 32; s <<= 1)
      tmax = fmaxf(tmax, __shfl_xor(tmax, s, 32));
    int done = __builtin_amdgcn_readfirstlane((int)(tmax <= TRANS_THR));
    if (done) break;
  }

  // D layout: VGPR r -> rows r (lanes 0-15) and r+8 (lanes 16-31), N=lane&15
#pragma unroll
  for (int r = 0; r < 8; ++r) {
    int row = hi ? r + 8 : r;
    dscr[wave][row * 16 + m] = acc[r];
  }
  __syncthreads();

  // 16 pixels x 3 channels = 48 coalesced stores per wave
  for (int t2 = lane; t2 < 48; t2 += 32) {
    int pp = t2 / 3, ch = t2 % 3;
    int gp = blockIdx.x * 128 + wave * 16 + pp;
    out[gp * 3 + ch] = dscr[wave][pp * 16 + ch];
  }
}

// ---------------------------------------------------------------------------
extern "C" void kernel_launch(void* const* d_in, const int* in_sizes, int n_in,
                              void* d_out, int out_size, void* d_ws,
                              size_t ws_size, hipStream_t stream) {
  const float* means  = (const float*)d_in[0];   // [1,512,2]
  const float* conics = (const float*)d_in[1];   // [1,512,3]
  const float* colors = (const float*)d_in[2];   // [1,512,3]
  const float* opac   = (const float*)d_in[3];   // [1,512]
  const float* depths = (const float*)d_in[4];   // [1,512]
  float* packed = (float*)d_ws;                  // 512*10 f32 = 20480 B
  float* out    = (float*)d_out;                 // [1,256,256,3]

  sort_pack_kernel<<<1, 512, 0, stream>>>(means, conics, colors, opac,
                                          depths, packed);
  // 65536 pixels / (8 waves * 16 pixels) = 512 blocks
  raster_kernel<<<512, 256, 0, stream>>>(packed, out);
}
